// EmbeddingBlock_44306882626082
// MI455X (gfx1250) — matrix operand order repaired and checked by reading, not compiled, not run on hardware
//
#include <hip/hip_runtime.h>
#include <hip/hip_bf16.h>

typedef _Float16 half16 __attribute__((ext_vector_type(16)));
typedef float    f32x8  __attribute__((ext_vector_type(8)));
typedef unsigned int u32x4 __attribute__((ext_vector_type(4)));
typedef int          i32x4 __attribute__((ext_vector_type(4)));
typedef int          i32x8 __attribute__((ext_vector_type(8)));

#define NTILES_N 8      // 128 outputs / 16
#define KSTEPS   12     // 384 / 32
#define HIDDEN   128
#define BLK      256    // 8 wave32 per block, 128 edges per block
#define B_BYTES  (NTILES_N * KSTEPS * 32 * 16 * 2)   // 98304 B = 96 KB

#if defined(__gfx1250__) && __has_builtin(__builtin_amdgcn_tensor_load_to_lds)
#define HAVE_TDM 1
#else
#define HAVE_TDM 0
#endif

// ---------------------------------------------------------------------------
// Kernel 0: convert lin_w [128 x 384] fp32 -> f16, pre-swizzled into the exact
// WMMA B-fragment layout (B is 32x16 KxN per step; lanes 0-15 hold K=0..7,16..23,
// lanes 16-31 hold K=8..15,24..31, halfs packed K-major within a lane).
// Linear id = (((nt*12 + kk)*32) + lane)*16 + (v*2 + h)
// ---------------------------------------------------------------------------
__global__ void convert_w_kernel(const float* __restrict__ lin_w,
                                 _Float16* __restrict__ bfrag) {
    int id = blockIdx.x * 256 + threadIdx.x;
    const int total = NTILES_N * KSTEPS * 32 * 16;
    if (id >= total) return;
    int sub  = id & 15;
    int lane = (id >> 4) & 31;
    int kk   = (id >> 9) % KSTEPS;
    int nt   = id / (16 * 32 * KSTEPS);
    int K = kk * 32 + ((lane >> 4) << 3) + sub + ((sub >> 3) << 3);
    int n = nt * 16 + (lane & 15);
    bfrag[id] = (_Float16)lin_w[n * 384 + K];     // B[k][n] = lin_w[n][k]
}

// ---------------------------------------------------------------------------
// Main fused kernel: gather + rbf-MLP + concat-GEMM (WMMA f16->f32) + silu.
// 256 threads = 8 wave32; each wave: 16 edges x 128 outputs.
// ---------------------------------------------------------------------------
__global__ __launch_bounds__(BLK)
void fused_edge_kernel(const int*   __restrict__ x,
                       const float* __restrict__ rbf,
                       const int*   __restrict__ gi,
                       const int*   __restrict__ gj,
                       const float* __restrict__ emb_w,
                       const float* __restrict__ lin_rbf_w,
                       const float* __restrict__ lin_rbf_b,
                       const float* __restrict__ lin_b,
                       const _Float16* __restrict__ bfrag,
                       float* __restrict__ out,
                       int E) {
    // Weight matrix in B-fragment order: 8 N-tiles x 12 K-steps x 32 lanes (96 KB)
    __shared__ half16 sB[NTILES_N * KSTEPS * 32];
    __shared__ float  sRW[HIDDEN * 6];   // lin_rbf_w  (row-major [c][r])
    __shared__ float  sRbfB[HIDDEN];     // lin_rbf_b
    __shared__ float  sLinB[HIDDEN];     // lin_b

    const int lane = threadIdx.x & 31;
    const int wave = threadIdx.x >> 5;

    // ---- B stage: one TDM descriptor moves 96 KB global->LDS (wave 0) ----
#if HAVE_TDM
    if (wave == 0) {
        const unsigned long long ga = (unsigned long long)bfrag;
        const unsigned int lds = (unsigned int)(unsigned long long)(void*)sB;
        // D# group 0: count=1 | lds_addr | global_addr | type=2
        u32x4 g0 = { 1u, lds, (unsigned int)ga,
                     (unsigned int)((ga >> 32) & 0x01FFFFFFu) | 0x80000000u };
        // D# group 1: data_size=8B (code 3); 1-D tile: dim0 = tile0 = 12288
        const unsigned int nelem = B_BYTES / 8;          // 12288
        i32x8 g1 = { (int)(3u << 16),                    // data_size
                     (int)((nelem & 0xFFFFu) << 16),     // tensor_dim0 lo
                     (int)(1u << 16),                    // dim0 hi=0 | tensor_dim1=1
                     (int)((nelem & 0xFFFFu) << 16),     // dim1 hi=0 | tile_dim0
                     0,                                  // tile_dim1/2 = unused
                     (int)nelem,                         // tensor_dim0_stride lo
                     0, 0 };
        i32x4 z4 = { 0, 0, 0, 0 };
#if __clang_major__ >= 23
        i32x8 z8 = { 0, 0, 0, 0, 0, 0, 0, 0 };
        __builtin_amdgcn_tensor_load_to_lds(g0, g1, z4, z4, z8, 0);
#else
        __builtin_amdgcn_tensor_load_to_lds(g0, g1, z4, z4, 0);
#endif
        __builtin_amdgcn_s_wait_tensorcnt(0);
    }
#else
    {
        const uint4* src = (const uint4*)bfrag;
        uint4*       dst = (uint4*)sB;
        for (int t = threadIdx.x; t < (int)(B_BYTES / 16); t += BLK) dst[t] = src[t];
    }
#endif
    // ---- small-table staging (other waves overlap with the TDM) ----
    for (int t = threadIdx.x; t < HIDDEN * 6; t += BLK) sRW[t] = lin_rbf_w[t];
    if (threadIdx.x >= BLK - HIDDEN) {
        int c = threadIdx.x - (BLK - HIDDEN);
        sRbfB[c] = lin_rbf_b[c];
        sLinB[c] = lin_b[c];
    }
    __syncthreads();

    const int m  = lane & 15;            // A-fragment row (edge within tile)
    const int hi = lane >> 4;            // lane-half selector for K groups
    const int tilebase = (blockIdx.x * 8 + wave) * 16;

    int e = tilebase + m;
    if (e >= E) e = E - 1;               // clamp loads; stores guarded below

    const int ii = gi[e];
    const int jj = gj[e];
    const float* rowi = emb_w + (long)x[ii] * HIDDEN;
    const float* rowj = emb_w + (long)x[jj] * HIDDEN;

    float r[6];
#pragma unroll
    for (int q = 0; q < 6; ++q) r[q] = rbf[(long)e * 6 + q];

    f32x8 acc[NTILES_N];
#pragma unroll
    for (int nt = 0; nt < NTILES_N; ++nt) acc[nt] = (f32x8)0.0f;

    // ---- phase 1+2: h[i] (kk 0..3) then h[j] (kk 4..7) ----
#pragma unroll 2
    for (int kk = 0; kk < 8; ++kk) {
        const float* src = (kk < 4) ? rowi : rowj;
        const int base = (kk & 3) * 32 + hi * 8;
        const float4* p = (const float4*)(src + base);
        float4 a0 = p[0], a1 = p[1];     // K base..base+7   -> halfs 0..7
        float4 a2 = p[4], a3 = p[5];     // K base+16..+23   -> halfs 8..15
        half16 A;
        A[0]=(_Float16)a0.x; A[1]=(_Float16)a0.y; A[2]=(_Float16)a0.z; A[3]=(_Float16)a0.w;
        A[4]=(_Float16)a1.x; A[5]=(_Float16)a1.y; A[6]=(_Float16)a1.z; A[7]=(_Float16)a1.w;
        A[8]=(_Float16)a2.x; A[9]=(_Float16)a2.y; A[10]=(_Float16)a2.z; A[11]=(_Float16)a2.w;
        A[12]=(_Float16)a3.x; A[13]=(_Float16)a3.y; A[14]=(_Float16)a3.z; A[15]=(_Float16)a3.w;
#pragma unroll
        for (int nt = 0; nt < NTILES_N; ++nt) {
            half16 B = sB[(nt * KSTEPS + kk) * 32 + lane];
            acc[nt] = __builtin_amdgcn_wmma_f32_16x16x32_f16(
                false, A, false, B, (short)0, acc[nt], false, false);
        }
    }

    // ---- phase 3: rbf MLP features (kk 8..11), fp32 math then cvt ----
#pragma unroll 2
    for (int kk = 8; kk < 12; ++kk) {
        const int cb = (kk - 8) * 32 + hi * 8;
        half16 A;
#pragma unroll
        for (int t = 0; t < 8; ++t) {
            int c0 = cb + t;
            int c1 = cb + 16 + t;
            float s0 = sRbfB[c0];
            float s1 = sRbfB[c1];
#pragma unroll
            for (int q = 0; q < 6; ++q) {
                s0 = fmaf(r[q], sRW[c0 * 6 + q], s0);
                s1 = fmaf(r[q], sRW[c1 * 6 + q], s1);
            }
            s0 = s0 / (1.0f + __expf(-s0));   // silu
            s1 = s1 / (1.0f + __expf(-s1));
            A[t]     = (_Float16)s0;
            A[t + 8] = (_Float16)s1;
        }
#pragma unroll
        for (int nt = 0; nt < NTILES_N; ++nt) {
            half16 B = sB[(nt * KSTEPS + kk) * 32 + lane];
            acc[nt] = __builtin_amdgcn_wmma_f32_16x16x32_f16(
                false, A, false, B, (short)0, acc[nt], false, false);
        }
    }

    // ---- epilogue: + lin_b, silu, store fp32 ----
    // C/D layout: lane -> N = lane&15 ; VGPR rr -> M = rr + 8*(lane>=16)
    const int ncol = m;
    const int ebase = tilebase + hi * 8;
#pragma unroll
    for (int nt = 0; nt < NTILES_N; ++nt) {
        const int n = nt * 16 + ncol;
        const float bias = sLinB[n];
#pragma unroll
        for (int rr = 0; rr < 8; ++rr) {
            const int eo = ebase + rr;
            if (eo < E) {
                float v = acc[nt][rr] + bias;
                v = v / (1.0f + __expf(-v));
                out[(long)eo * HIDDEN + n] = v;
            }
        }
    }
}

// ---------------------------------------------------------------------------
extern "C" void kernel_launch(void* const* d_in, const int* in_sizes, int n_in,
                              void* d_out, int out_size, void* d_ws, size_t ws_size,
                              hipStream_t stream) {
    const int*   x         = (const int*)d_in[0];
    const float* rbf       = (const float*)d_in[1];
    const int*   gi        = (const int*)d_in[2];
    const int*   gj        = (const int*)d_in[3];
    const float* emb_w     = (const float*)d_in[4];
    const float* lin_rbf_w = (const float*)d_in[5];
    const float* lin_rbf_b = (const float*)d_in[6];
    const float* lin_w     = (const float*)d_in[7];
    const float* lin_b     = (const float*)d_in[8];
    float*       out       = (float*)d_out;
    const int E = in_sizes[2];                 // edge count (size of i)

    _Float16* bfrag = (_Float16*)d_ws;         // 96 KB of workspace
    const int nfrag = NTILES_N * KSTEPS * 32 * 16;
    convert_w_kernel<<<(nfrag + 255) / 256, 256, 0, stream>>>(lin_w, bfrag);

    const int nblocks = (E + 127) / 128;       // 128 edges per block (8 waves)
    fused_edge_kernel<<<nblocks, BLK, 0, stream>>>(
        x, rbf, gi, gj, emb_w, lin_rbf_w, lin_rbf_b, lin_b, bfrag, out, E);
}
